// RegionFusionNetwork_43628277793176
// MI455X (gfx1250) — compile-verified, matrix operand order checked by reading, not run on hardware
//
#include <hip/hip_runtime.h>
#include <hip/hip_bf16.h>
#include <cstdint>

typedef __attribute__((ext_vector_type(16))) _Float16 v16h;
typedef __attribute__((ext_vector_type(8)))  float    v8f;

#define NROI   2000
#define MT16   126          // M tiles (2000 rows -> 125 real tiles + 1 zero pad tile)
#define MWT    63           // wave rows (each wave covers 2 M-tiles = 32 rows)
#define CCH    128
#define DFEAT  6272         // 128 * 49
#define KS1    588          // 3 * 6272 / 32  (fused 3-modality K)
#define KS2    16           // 512 / 32
#define NT     32           // 512 / 16
#define SDIM   512
#define FROWS  2016         // padded rows for f0/f1 (126 * 16)

// ---- WMMA 16-bit A/B fragment k-index mapping (ISA 7.12.2, 16-bit A 16x32) ----
// lanes 0-15 (h=0): v0:K0,1 v1:K2,3 v2:K4,5 v3:K6,7 v4:K16,17 .. v7:K22,23
// lanes 16-31(h=1): v0:K8,9 .. v3:K14,15 v4:K24,25 .. v7:K30,31
__device__ __forceinline__ int kin_from_e(int e, int h) {
    int vg = e >> 1, bit = e & 1;
    int base = (vg < 4) ? (2 * vg) : (16 + 2 * (vg - 4));
    return base + 8 * h + bit;
}
__device__ __forceinline__ int e_from_kin(int kin, int* hout) {
    int h = (kin >> 3) & 1;
    int base = kin - 8 * h;                      // in {0..7} U {16..23}
    int vg = (base < 8) ? (base >> 1) : (4 + ((base - 16) >> 1));
    *hout = h;
    return 2 * vg + (base & 1);
}

// ---- ROI Align -> directly packed WMMA A-fragments (f16) ----
__global__ void roi_pack_kernel(const float* __restrict__ feat, int H, int W,
                                const float* __restrict__ p3d,
                                const float* __restrict__ rnd, float rscale,
                                int mode, int kbase, _Float16* __restrict__ pA)
{
    __shared__ int   s_y0[14], s_y1[14], s_x0[14], s_x1[14];
    __shared__ float s_ly[14], s_lx[14];
    __shared__ int   s_vy[14], s_vx[14];

    const int n   = blockIdx.x;
    const int tid = threadIdx.x;

    const int b = (int)p3d[(size_t)n * 8 + 0];
    float x1, y1, x2, y2;
    if (mode == 0) {                                  // project_bev: X0=0, Y0=-40, VOX=0.1
        float xi = p3d[(size_t)n * 8 + 1] * 10.0f;
        float yi = (p3d[(size_t)n * 8 + 2] + 40.0f) * 10.0f;
        float li = p3d[(size_t)n * 8 + 4] * 10.0f;
        float wi = p3d[(size_t)n * 8 + 5] * 10.0f;
        x1 = xi - 0.5f * li; x2 = xi + 0.5f * li;
        y1 = yi - 0.5f * wi; y2 = yi + 0.5f * wi;
    } else {
        x1 = rnd[(size_t)n * 4 + 0] * rscale; y1 = rnd[(size_t)n * 4 + 1] * rscale;
        x2 = rnd[(size_t)n * 4 + 2] * rscale; y2 = rnd[(size_t)n * 4 + 3] * rscale;
    }
    float roi_w = fmaxf(x2 - x1, 1.0f), roi_h = fmaxf(y2 - y1, 1.0f);
    float bin_h = roi_h * (1.0f / 7.0f), bin_w = roi_w * (1.0f / 7.0f);

    if (tid < 14) {                                   // y sample geometry
        int s = tid;
        float g = y1 + (s + 0.5f) * 0.5f * bin_h;
        s_vy[s] = (g >= -1.0f) && (g <= (float)H);
        float yc = fminf(fmaxf(g, 0.0f), (float)(H - 1));
        int y0 = (int)floorf(yc);
        s_y0[s] = y0; s_y1[s] = min(y0 + 1, H - 1); s_ly[s] = yc - (float)y0;
    } else if (tid >= 16 && tid < 30) {               // x sample geometry
        int s = tid - 16;
        float g = x1 + (s + 0.5f) * 0.5f * bin_w;
        s_vx[s] = (g >= -1.0f) && (g <= (float)W);
        float xc = fminf(fmaxf(g, 0.0f), (float)(W - 1));
        int x0 = (int)floorf(xc);
        s_x0[s] = x0; s_x1[s] = min(x0 + 1, W - 1); s_lx[s] = xc - (float)x0;
    }
    __syncthreads();

    const int mt  = n >> 4;
    const int mlo = n & 15;
    for (int i = tid; i < DFEAT; i += blockDim.x) {
        int c = i / 49, p = i % 49;
        int oy = p / 7, ox = p % 7;
        const float* f = feat + (size_t)(b * CCH + c) * H * W;
        float acc = 0.0f;
        #pragma unroll
        for (int sy = 0; sy < 2; ++sy) {
            int ys = 2 * oy + sy;
            #pragma unroll
            for (int sx = 0; sx < 2; ++sx) {
                int xs = 2 * ox + sx;
                if (s_vy[ys] && s_vx[xs]) {
                    float ly = s_ly[ys], lx = s_lx[xs];
                    const float* r0 = f + (size_t)s_y0[ys] * W;
                    const float* r1 = f + (size_t)s_y1[ys] * W;
                    float v00 = r0[s_x0[xs]], v01 = r0[s_x1[xs]];
                    float v10 = r1[s_x0[xs]], v11 = r1[s_x1[xs]];
                    acc += (1.0f - ly) * ((1.0f - lx) * v00 + lx * v01)
                         +          ly * ((1.0f - lx) * v10 + lx * v11);
                }
            }
        }
        acc *= 0.25f;                                  // mean over 2x2 samples
        int k = kbase + i;
        int ks = k >> 5, kin = k & 31;
        int h, e;
        e = e_from_kin(kin, &h);
        pA[(((size_t)mt * KS1 + ks) * 32 + (mlo + (h << 4))) * 16 + e] = (_Float16)acc;
    }
}

// ---- Zero the pad M-tile slab of packed A1 (workspace is poisoned) ----
__global__ void zero_pad_kernel(_Float16* __restrict__ pA)
{
    size_t i = (size_t)blockIdx.x * blockDim.x + threadIdx.x;
    size_t total = (size_t)KS1 * 32 * 16;            // one M-tile slab
    if (i < total) pA[(size_t)125 * KS1 * 32 * 16 + i] = (_Float16)0.0f;
}

// ---- Pack weight matrix W[n][k] (row-major, n=out col) into B fragments ----
__global__ void pack_w_kernel(const float* __restrict__ Wm, int Kdim,
                              int ks_total, int ks_off, int ks_local,
                              _Float16* __restrict__ pB)
{
    int t = blockIdx.x * blockDim.x + threadIdx.x;
    if (t >= NT * ks_local * 32) return;
    int lane = t & 31;
    int ksl  = (t >> 5) % ks_local;
    int nt   = t / (32 * ks_local);
    int nrow = nt * 16 + (lane & 15);
    int h    = lane >> 4;
    _Float16* dst = pB + (((size_t)nt * ks_total + ks_off + ksl) * 32 + lane) * 16;
    #pragma unroll
    for (int e = 0; e < 16; ++e) {
        int kin = kin_from_e(e, h);
        dst[e] = (_Float16)Wm[(size_t)nrow * Kdim + (ksl * 32 + kin)];
    }
}

// ---- Pack f0 (row-major f32) into A fragments for the fusion GEMM ----
__global__ void pack_a2_kernel(const float* __restrict__ f0, _Float16* __restrict__ pA)
{
    int t = blockIdx.x * blockDim.x + threadIdx.x;
    if (t >= MT16 * KS2 * 32) return;
    int lane = t & 31;
    int ks = (t >> 5) % KS2;
    int mt = t / (32 * KS2);
    int m  = mt * 16 + (lane & 15);
    int h  = lane >> 4;
    _Float16* dst = pA + (((size_t)mt * KS2 + ks) * 32 + lane) * 16;
    #pragma unroll
    for (int e = 0; e < 16; ++e) {
        int kin = kin_from_e(e, h);
        float v = (m < NROI) ? f0[(size_t)m * SDIM + ks * 32 + kin] : 0.0f;
        dst[e] = (_Float16)v;
    }
}

__global__ void bias3_kernel(const float* __restrict__ a, const float* __restrict__ b,
                             const float* __restrict__ c, float* __restrict__ o)
{
    int i = blockIdx.x * blockDim.x + threadIdx.x;
    if (i < SDIM) o[i] = a[i] + b[i] + c[i];
}

// ---- WMMA GEMM: each wave computes a 32x64 tile (2x4 WMMA tiles, 8 accums) ----
// KS is a template parameter so the inter-tile stride (KS*1024 B, max 602112 B
// < 2^23) folds into the 24-bit instruction offset: only TWO live 64-bit
// pointers, 12 immediate-offset b128 loads and 8 WMMAs per k-step, 2 pointer
// bumps. __launch_bounds__(256,1) lifts the VGPR cap so nothing spills.
template<int KS>
__global__ __launch_bounds__(256, 1)
void wmma_gemm_kernel(const _Float16* __restrict__ pA,
                      const _Float16* __restrict__ pB,
                      const float* __restrict__ bias, float scale,
                      float* __restrict__ out)
{
    const int wave = (blockIdx.x * blockDim.x + threadIdx.x) >> 5;
    const int lane = threadIdx.x & 31;
    const int wrow = wave >> 3;          // 0..62 : block's 8 waves share wrow
    const int wcol = wave & 7;           // 0..7
    if (wrow >= MWT) return;
    const int mt0 = wrow * 2;
    const int nt0 = wcol * 4;

    constexpr size_t TS = (size_t)KS * 32 * 16;   // elements per whole tile slab
    const _Float16* ap = pA + (((size_t)mt0 * KS) * 32 + lane) * 16;
    const _Float16* bp = pB + (((size_t)nt0 * KS) * 32 + lane) * 16;

    v8f acc[2][4] = {};
    for (int ks = 0; ks < KS; ++ks) {
        v16h a0 = *(const v16h*)(ap);
        v16h a1 = *(const v16h*)(ap + TS);
        v16h b0 = *(const v16h*)(bp);
        v16h b1 = *(const v16h*)(bp + TS);
        v16h b2 = *(const v16h*)(bp + 2 * TS);
        v16h b3 = *(const v16h*)(bp + 3 * TS);
        acc[0][0] = __builtin_amdgcn_wmma_f32_16x16x32_f16(false, a0, false, b0, (short)0, acc[0][0], false, false);
        acc[0][1] = __builtin_amdgcn_wmma_f32_16x16x32_f16(false, a0, false, b1, (short)0, acc[0][1], false, false);
        acc[0][2] = __builtin_amdgcn_wmma_f32_16x16x32_f16(false, a0, false, b2, (short)0, acc[0][2], false, false);
        acc[0][3] = __builtin_amdgcn_wmma_f32_16x16x32_f16(false, a0, false, b3, (short)0, acc[0][3], false, false);
        acc[1][0] = __builtin_amdgcn_wmma_f32_16x16x32_f16(false, a1, false, b0, (short)0, acc[1][0], false, false);
        acc[1][1] = __builtin_amdgcn_wmma_f32_16x16x32_f16(false, a1, false, b1, (short)0, acc[1][1], false, false);
        acc[1][2] = __builtin_amdgcn_wmma_f32_16x16x32_f16(false, a1, false, b2, (short)0, acc[1][2], false, false);
        acc[1][3] = __builtin_amdgcn_wmma_f32_16x16x32_f16(false, a1, false, b3, (short)0, acc[1][3], false, false);
        ap += 32 * 16;                   // advance one k-step within the slab
        bp += 32 * 16;
    }

    // C/D layout: vgpr r -> M = r (+8 for lanes 16-31), lane&15 -> N
    const int nlo  = lane & 15;
    const int mofs = (lane >> 4) << 3;
    #pragma unroll
    for (int i = 0; i < 2; ++i) {
        #pragma unroll
        for (int j = 0; j < 4; ++j) {
            int ncol = (nt0 + j) * 16 + nlo;
            float bs = bias[ncol];
            #pragma unroll
            for (int r = 0; r < 8; ++r) {
                int m = (mt0 + i) * 16 + r + mofs;
                out[(size_t)m * SDIM + ncol] = (acc[i][j][r] + bs) * scale;
            }
        }
    }
}

// ---- Tiny cls/reg heads (f32, precision-safe, compute-trivial) ----
__global__ void heads_kernel(const float* __restrict__ f1,
                             const float* __restrict__ Wc, const float* __restrict__ bc,
                             const float* __restrict__ Wr, const float* __restrict__ br,
                             float* __restrict__ out)
{
    int t = blockIdx.x * blockDim.x + threadIdx.x;
    if (t >= NROI * 26) return;
    int n = t / 26, j = t % 26;
    const float* x = f1 + (size_t)n * SDIM;
    const float* w; float bias; float* o;
    if (j < 2) { w = Wc + (size_t)j * SDIM; bias = bc[j]; o = out + (size_t)n * 2 + j; }
    else { int jr = j - 2; w = Wr + (size_t)jr * SDIM; bias = br[jr];
           o = out + (size_t)NROI * 2 + (size_t)n * 24 + jr; }
    float acc = bias;
    #pragma unroll 8
    for (int k = 0; k < SDIM; ++k) acc += x[k] * w[k];
    *o = acc;
}

extern "C" void kernel_launch(void* const* d_in, const int* in_sizes, int n_in,
                              void* d_out, int out_size, void* d_ws, size_t ws_size,
                              hipStream_t stream)
{
    (void)in_sizes; (void)n_in; (void)out_size; (void)ws_size;
    const float* feat_bev = (const float*)d_in[0];
    const float* feat_fv  = (const float*)d_in[1];
    const float* feat_rgb = (const float*)d_in[2];
    const float* p3d      = (const float*)d_in[3];
    const float* rand_fv  = (const float*)d_in[4];
    const float* rand_rgb = (const float*)d_in[5];
    const float* W_bev = (const float*)d_in[6];  const float* b_bev = (const float*)d_in[7];
    const float* W_fv  = (const float*)d_in[8];  const float* b_fv  = (const float*)d_in[9];
    const float* W_rgb = (const float*)d_in[10]; const float* b_rgb = (const float*)d_in[11];
    const float* W_fus = (const float*)d_in[12]; const float* b_fus = (const float*)d_in[13];
    const float* W_cls = (const float*)d_in[14]; const float* b_cls = (const float*)d_in[15];
    const float* W_reg = (const float*)d_in[16]; const float* b_reg = (const float*)d_in[17];
    float* out = (float*)d_out;

    // Workspace carve-out (256B aligned; total ~106 MB)
    char* ws = (char*)d_ws;
    size_t off = 0;
    auto carve = [&](size_t bytes) {
        char* p = ws + off;
        off = (off + bytes + 255) & ~(size_t)255;
        return p;
    };
    _Float16* pA1 = (_Float16*)carve((size_t)MT16 * KS1 * 32 * 16 * sizeof(_Float16)); // 75.9 MB
    _Float16* pB1 = (_Float16*)carve((size_t)NT   * KS1 * 32 * 16 * sizeof(_Float16)); // 19.3 MB
    _Float16* pB2 = (_Float16*)carve((size_t)NT   * KS2 * 32 * 16 * sizeof(_Float16));
    _Float16* pA2 = (_Float16*)carve((size_t)MT16 * KS2 * 32 * 16 * sizeof(_Float16));
    float* f0   = (float*)carve((size_t)FROWS * SDIM * sizeof(float));
    float* f1   = (float*)carve((size_t)FROWS * SDIM * sizeof(float));
    float* bsum = (float*)carve(SDIM * sizeof(float));

    // 1) Pack weights into B fragments (3 big FCs share one fused-K buffer)
    int wthreads = NT * 196 * 32;
    pack_w_kernel<<<(wthreads + 255) / 256, 256, 0, stream>>>(W_bev, DFEAT, KS1, 0,   196, pB1);
    pack_w_kernel<<<(wthreads + 255) / 256, 256, 0, stream>>>(W_fv,  DFEAT, KS1, 196, 196, pB1);
    pack_w_kernel<<<(wthreads + 255) / 256, 256, 0, stream>>>(W_rgb, DFEAT, KS1, 392, 196, pB1);
    int w2threads = NT * KS2 * 32;
    pack_w_kernel<<<(w2threads + 255) / 256, 256, 0, stream>>>(W_fus, SDIM, KS2, 0, KS2, pB2);

    bias3_kernel<<<2, 256, 0, stream>>>(b_bev, b_fv, b_rgb, bsum);

    // 2) Zero the pad M-tile, then ROI-align straight into packed A fragments
    {
        int zthreads = KS1 * 32 * 16;
        zero_pad_kernel<<<(zthreads + 255) / 256, 256, 0, stream>>>(pA1);
    }
    roi_pack_kernel<<<NROI, 256, 0, stream>>>(feat_bev, 800, 704, p3d, nullptr,  0.0f,   0, 0,         pA1);
    roi_pack_kernel<<<NROI, 256, 0, stream>>>(feat_fv,   64, 512, p3d, rand_fv,  50.0f,  1, DFEAT,     pA1);
    roi_pack_kernel<<<NROI, 256, 0, stream>>>(feat_rgb, 128, 512, p3d, rand_rgb, 100.0f, 1, 2 * DFEAT, pA1);

    // 3) Fused 3-modality GEMM: f0 = (A_bev Wb^T + A_fv Wf^T + A_rgb Wr^T + b_sum)/3
    wmma_gemm_kernel<KS1><<<MWT, 256, 0, stream>>>(pA1, pB1, bsum, 1.0f / 3.0f, f0);

    // 4) Fusion FC: f1 = f0 @ W_fus^T + b_fus
    pack_a2_kernel<<<(MT16 * KS2 * 32 + 255) / 256, 256, 0, stream>>>(f0, pA2);
    wmma_gemm_kernel<KS2><<<MWT, 256, 0, stream>>>(pA2, pB2, b_fus, 1.0f, f1);

    // 5) Heads -> d_out = [cls (2000x2) | reg (2000x24)]
    heads_kernel<<<(NROI * 26 + 255) / 256, 256, 0, stream>>>(f1, W_cls, b_cls, W_reg, b_reg, out);
}
